// GroupedMLP_43731357008245
// MI455X (gfx1250) — matrix operand order repaired and checked by reading, not compiled, not run on hardware
//
#include <hip/hip_runtime.h>
#include <hip/hip_bf16.h>

// Problem constants (match reference)
#define NEXP    8
#define HID     2048
#define FFN     8192
#define TPE     1024      // tokens per expert (balanced)
#define TWO_FFN 16384

typedef __attribute__((ext_vector_type(16))) __bf16 v16bf;
typedef __attribute__((ext_vector_type(8)))  __bf16 v8bf;
typedef __attribute__((ext_vector_type(8)))  float  v8f;
typedef __attribute__((ext_vector_type(4)))  float  v4f;

union FragAB { v16bf v; struct { v8bf lo, hi; } h; };

__device__ __forceinline__ unsigned pack_bf2(float a, float b) {
    union { __bf16 h[2]; unsigned u; } p;
    p.h[0] = (__bf16)a; p.h[1] = (__bf16)b;
    return p.u;
}

// LDS row stride in halves: 32 K-values + 8 pad -> 80B rows (16B aligned,
// 20-dword stride => the 16 lanes of a b128 read tile all 64 banks).
#define LDS_K 40

#if __has_builtin(__builtin_amdgcn_global_load_async_to_lds_b128)
#define HAS_ASYNC_LDS 1
typedef __attribute__((ext_vector_type(4))) int v4i;
typedef __attribute__((address_space(1))) v4i gbl_v4i;
typedef __attribute__((address_space(3))) v4i lds_v4i;
#else
#define HAS_ASYNC_LDS 0
#endif

__device__ __forceinline__ void wait_async0() {
#if __has_builtin(__builtin_amdgcn_s_wait_asynccnt)
    __builtin_amdgcn_s_wait_asynccnt(0);
#else
    asm volatile("s_wait_asynccnt 0x0" ::: "memory");
#endif
}

// ---------------------------------------------------------------------------
// Kernel 1: act[e,t,j] = silu(x@w1_gate) * (x@w1_val), stored bf16 in ws.
// Block: 256 thr = 8 waves. Block tile: 128 tokens x 64 act cols (128 staged
// w1 cols, interleaved [gate 0-31 | val 0-31 | gate 32-63 | val 32-63]).
// Wave tile: 32 rows x 64 staged cols = 2 A-frags x 4 B-frags -> 8 WMMA with
// only 12 ds_load_b128 (1.5 per WMMA). K over HID in steps of 32, dbl-buffered.
// ---------------------------------------------------------------------------
__global__ __launch_bounds__(256)
void fc1_swiglu_kernel(const float* __restrict__ x,
                       const float* __restrict__ w1,
                       __bf16* __restrict__ act) {
    __shared__ __align__(16) __bf16 sX[2][128 * LDS_K];
    __shared__ __align__(16) __bf16 sW[2][128 * LDS_K];

    const int tid  = threadIdx.x;
    const int lane = tid & 31;
    const int wave = tid >> 5;
    const int hf   = lane >> 4;     // 0: lanes 0-15, 1: lanes 16-31
    const int ln16 = lane & 15;
    const int rg   = wave >> 1;     // row group: 32 tokens each
    const int cg   = wave & 1;      // col group: 64 staged cols each

    const int e  = blockIdx.z;
    const int t0 = blockIdx.y * 128;
    const int n0 = blockIdx.x * 64;

    const float* Xg  = x  + ((size_t)e * TPE + t0) * HID;
    const float* W1e = w1 + (size_t)e * HID * TWO_FFN;

    v8f acc[8];   // acc[a2*4 + c]: a2 = row sub-tile (0/1), c = col tile (0..3)
#pragma unroll
    for (int c = 0; c < 8; ++c)
#pragma unroll
        for (int r = 0; r < 8; ++r) acc[c][r] = 0.0f;

    v4f xr[4], wr[4];

    // Staged col sc bits: [6]=cg-block, [5]=val flag, [4:0]=col-in-32.
#define K1_LOAD_REGS(KK)                                                        \
    do {                                                                        \
        _Pragma("unroll")                                                       \
        for (int r = 0; r < 4; ++r) {                                           \
            int j = tid + 256 * r;                                              \
            int m = j >> 3, q = j & 7;                                          \
            xr[r] = *(const v4f*)&Xg[(size_t)m * HID + (KK) + q * 4];           \
            int k = j >> 5, c4 = j & 31;                                        \
            int sc0  = c4 * 4;                                                  \
            int gcol = ((sc0 & 32) ? FFN : 0) + n0 + ((sc0 >> 6) * 32)          \
                       + (sc0 & 31);                                            \
            wr[r] = *(const v4f*)&W1e[(size_t)((KK) + k) * TWO_FFN + gcol];     \
        }                                                                       \
    } while (0)

#define K1_STORE_LDS(BUF)                                                       \
    do {                                                                        \
        _Pragma("unroll")                                                       \
        for (int r = 0; r < 4; ++r) {                                           \
            int j = tid + 256 * r;                                              \
            int m = j >> 3, q = j & 7;                                          \
            *(uint2*)&sX[BUF][m * LDS_K + q * 4] =                              \
                make_uint2(pack_bf2(xr[r].x, xr[r].y),                          \
                           pack_bf2(xr[r].z, xr[r].w));                         \
            int k = j >> 5, c4 = j & 31;                                        \
            int nc = c4 * 4;                                                    \
            sW[BUF][(nc + 0) * LDS_K + k] = (__bf16)wr[r].x;                    \
            sW[BUF][(nc + 1) * LDS_K + k] = (__bf16)wr[r].y;                    \
            sW[BUF][(nc + 2) * LDS_K + k] = (__bf16)wr[r].z;                    \
            sW[BUF][(nc + 3) * LDS_K + k] = (__bf16)wr[r].w;                    \
        }                                                                       \
    } while (0)

    K1_LOAD_REGS(0);
    K1_STORE_LDS(0);

    const int NS = HID / 32;
    for (int s = 0; s < NS; ++s) {
        __syncthreads();
        const int cur = s & 1;
        const bool more = (s + 1) < NS;
        if (more) K1_LOAD_REGS((s + 1) * 32);

        FragAB a0, a1;
        {
            const __bf16* ap = &sX[cur][(rg * 32 + ln16) * LDS_K + 8 * hf];
            a0.h.lo = *(const v8bf*)ap;
            a0.h.hi = *(const v8bf*)(ap + 16);
            ap += 16 * LDS_K;
            a1.h.lo = *(const v8bf*)ap;
            a1.h.hi = *(const v8bf*)(ap + 16);
        }
        FragAB b[4];
#pragma unroll
        for (int c = 0; c < 4; ++c) {
            const __bf16* bp =
                &sW[cur][((cg * 4 + c) * 16 + ln16) * LDS_K + 8 * hf];
            b[c].h.lo = *(const v8bf*)bp;
            b[c].h.hi = *(const v8bf*)(bp + 16);
        }
#pragma unroll
        for (int c = 0; c < 4; ++c)
            acc[c] = __builtin_amdgcn_wmma_f32_16x16x32_bf16(
                false, a0.v, false, b[c].v, (short)0, acc[c], false, false);
#pragma unroll
        for (int c = 0; c < 4; ++c)
            acc[4 + c] = __builtin_amdgcn_wmma_f32_16x16x32_bf16(
                false, a1.v, false, b[c].v, (short)0, acc[4 + c], false, false);

        if (more) K1_STORE_LDS((s + 1) & 1);
    }

    // SwiGLU epilogue: within each a2 group, tiles {0,1}=gate, {2,3}=val
    __bf16* actE = act + (size_t)e * TPE * FFN;
#pragma unroll
    for (int a2 = 0; a2 < 2; ++a2) {
#pragma unroll
        for (int cp = 0; cp < 2; ++cp) {
#pragma unroll
            for (int r = 0; r < 8; ++r) {
                float g = acc[a2 * 4 + cp][r];
                float v = acc[a2 * 4 + 2 + cp][r];
                float sg = g / (1.0f + __expf(-g));
                int t   = t0 + rg * 32 + a2 * 16 + r + 8 * hf;
                int col = n0 + cg * 32 + cp * 16 + ln16;
                actE[(size_t)t * FFN + col] = (__bf16)(sg * v);
            }
        }
    }
#undef K1_LOAD_REGS
#undef K1_STORE_LDS
}

// ---------------------------------------------------------------------------
// Kernel 2: out[e,t,h] = act(bf16) @ w2(fp32->bf16), f32 accumulate/output.
// Block tile: 128 tokens x 128 out cols; wave tile 32x64 (2 A x 4 B frags).
// A-tile (already bf16) staged via GLOBAL_LOAD_ASYNC_TO_LDS_B128 (ASYNCcnt
// path); W2 needs f32->bf16 so it goes through VGPRs.
// ---------------------------------------------------------------------------
__global__ __launch_bounds__(256)
void fc2_kernel(const __bf16* __restrict__ act,
                const float* __restrict__ w2,
                float* __restrict__ out) {
    __shared__ __align__(16) __bf16 sA[2][128 * LDS_K];
    __shared__ __align__(16) __bf16 sW[2][128 * LDS_K];

    const int tid  = threadIdx.x;
    const int lane = tid & 31;
    const int wave = tid >> 5;
    const int hf   = lane >> 4;
    const int ln16 = lane & 15;
    const int rg   = wave >> 1;
    const int cg   = wave & 1;

    const int e  = blockIdx.z;
    const int t0 = blockIdx.y * 128;
    const int h0 = blockIdx.x * 128;

    const __bf16* Ae  = act + (size_t)e * TPE * FFN + (size_t)t0 * FFN;
    const float*  W2e = w2 + (size_t)e * FFN * HID;

    v8f acc[8];
#pragma unroll
    for (int c = 0; c < 8; ++c)
#pragma unroll
        for (int r = 0; r < 8; ++r) acc[c][r] = 0.0f;

    v4f wr[4];
#if !HAS_ASYNC_LDS
    uint4 ar[2];
#endif

#if HAS_ASYNC_LDS
#define K2_STAGE_A(KK, BUF)                                                     \
    do {                                                                        \
        _Pragma("unroll")                                                       \
        for (int r = 0; r < 2; ++r) {                                           \
            int j = tid + 256 * r;                                              \
            int m = j >> 2, q = j & 3;                                          \
            __builtin_amdgcn_global_load_async_to_lds_b128(                     \
                (gbl_v4i*)(void*)&Ae[(size_t)m * FFN + (KK) + 8 * q],           \
                (lds_v4i*)(void*)&sA[BUF][m * LDS_K + 8 * q], 0, 0);            \
        }                                                                       \
    } while (0)
#else
#define K2_STAGE_A(KK, BUF) /* split: loads here, stores in K2_STORE_A */       \
    do {                                                                        \
        _Pragma("unroll")                                                       \
        for (int r = 0; r < 2; ++r) {                                           \
            int j = tid + 256 * r;                                              \
            int m = j >> 2, q = j & 3;                                          \
            ar[r] = *(const uint4*)&Ae[(size_t)m * FFN + (KK) + 8 * q];         \
        }                                                                       \
    } while (0)
#endif

#define K2_LOAD_W(KK)                                                           \
    do {                                                                        \
        _Pragma("unroll")                                                       \
        for (int r = 0; r < 4; ++r) {                                           \
            int j = tid + 256 * r;                                              \
            int k = j >> 5, c4 = j & 31;                                        \
            wr[r] = *(const v4f*)&W2e[(size_t)((KK) + k) * HID + h0 + c4 * 4];  \
        }                                                                       \
    } while (0)

#define K2_STORE_LDS(BUF)                                                       \
    do {                                                                        \
        _Pragma("unroll")                                                       \
        for (int r = 0; r < 4; ++r) {                                           \
            int j = tid + 256 * r;                                              \
            int k = j >> 5, c4 = j & 31;                                        \
            int nc = c4 * 4;                                                    \
            sW[BUF][(nc + 0) * LDS_K + k] = (__bf16)wr[r].x;                    \
            sW[BUF][(nc + 1) * LDS_K + k] = (__bf16)wr[r].y;                    \
            sW[BUF][(nc + 2) * LDS_K + k] = (__bf16)wr[r].z;                    \
            sW[BUF][(nc + 3) * LDS_K + k] = (__bf16)wr[r].w;                    \
        }                                                                       \
    } while (0)

#if !HAS_ASYNC_LDS
#define K2_STORE_A(BUF)                                                         \
    do {                                                                        \
        _Pragma("unroll")                                                       \
        for (int r = 0; r < 2; ++r) {                                           \
            int j = tid + 256 * r;                                              \
            int m = j >> 2, q = j & 3;                                          \
            *(uint4*)&sA[BUF][m * LDS_K + 8 * q] = ar[r];                       \
        }                                                                       \
    } while (0)
#else
#define K2_STORE_A(BUF) do { } while (0)
#endif

    // Prologue: stage step 0
    K2_STAGE_A(0, 0);
    K2_LOAD_W(0);
    K2_STORE_LDS(0);
    K2_STORE_A(0);
#if HAS_ASYNC_LDS
    wait_async0();
#endif

    const int NS = FFN / 32;
    for (int s = 0; s < NS; ++s) {
        __syncthreads();
        const int cur = s & 1;
        const bool more = (s + 1) < NS;
        if (more) {
            K2_STAGE_A((s + 1) * 32, (s + 1) & 1);  // async: fire & forget here
            K2_LOAD_W((s + 1) * 32);
        }

        FragAB a0, a1;
        {
            const __bf16* ap = &sA[cur][(rg * 32 + ln16) * LDS_K + 8 * hf];
            a0.h.lo = *(const v8bf*)ap;
            a0.h.hi = *(const v8bf*)(ap + 16);
            ap += 16 * LDS_K;
            a1.h.lo = *(const v8bf*)ap;
            a1.h.hi = *(const v8bf*)(ap + 16);
        }
        FragAB b[4];
#pragma unroll
        for (int c = 0; c < 4; ++c) {
            const __bf16* bp =
                &sW[cur][((cg * 4 + c) * 16 + ln16) * LDS_K + 8 * hf];
            b[c].h.lo = *(const v8bf*)bp;
            b[c].h.hi = *(const v8bf*)(bp + 16);
        }
#pragma unroll
        for (int c = 0; c < 4; ++c)
            acc[c] = __builtin_amdgcn_wmma_f32_16x16x32_bf16(
                false, a0.v, false, b[c].v, (short)0, acc[c], false, false);
#pragma unroll
        for (int c = 0; c < 4; ++c)
            acc[4 + c] = __builtin_amdgcn_wmma_f32_16x16x32_bf16(
                false, a1.v, false, b[c].v, (short)0, acc[4 + c], false, false);

        if (more) {
            K2_STORE_LDS((s + 1) & 1);
            K2_STORE_A((s + 1) & 1);
#if HAS_ASYNC_LDS
            wait_async0();   // own async writes visible before barrier signal
#endif
        }
    }

    float* outE = out + ((size_t)e * TPE + t0) * HID;
#pragma unroll
    for (int a2 = 0; a2 < 2; ++a2) {
#pragma unroll
        for (int c = 0; c < 4; ++c) {
#pragma unroll
            for (int r = 0; r < 8; ++r) {
                int t    = rg * 32 + a2 * 16 + r + 8 * hf;
                int hcol = h0 + cg * 64 + c * 16 + ln16;
                outE[(size_t)t * HID + hcol] = acc[a2 * 4 + c][r];
            }
        }
    }
#undef K2_STAGE_A
#undef K2_LOAD_W
#undef K2_STORE_LDS
#undef K2_STORE_A
}

// ---------------------------------------------------------------------------
extern "C" void kernel_launch(void* const* d_in, const int* in_sizes, int n_in,
                              void* d_out, int out_size, void* d_ws, size_t ws_size,
                              hipStream_t stream) {
    const float* x  = (const float*)d_in[0];   // [8192, 2048] f32
    // d_in[1] = tokens_per_expert (static balanced -> unused)
    const float* w1 = (const float*)d_in[2];   // [8, 2048, 16384] f32
    const float* w2 = (const float*)d_in[3];   // [8, 8192, 2048] f32
    float* out = (float*)d_out;                // [8192, 2048] f32

    // Workspace: bf16 activations [E, TPE, FFN] = 128 MiB
    __bf16* act = (__bf16*)d_ws;

    dim3 blk(256);
    // grid.x fastest over FFN tiles so all tiles of one expert are co-resident
    // -> w1/w2 of each expert (134/67 MB) stream HBM once, reused via 192MB L2.
    dim3 g1(FFN / 64, TPE / 128, NEXP);
    fc1_swiglu_kernel<<<g1, blk, 0, stream>>>(x, w1, act);

    dim3 g2(HID / 128, TPE / 128, NEXP);
    fc2_kernel<<<g2, blk, 0, stream>>>(act, w2, out);
}
